// CompositionScorerNet_19499151524542
// MI455X (gfx1250) — matrix operand order repaired and checked by reading, not compiled, not run on hardware
//
#include <hip/hip_runtime.h>
#include <hip/hip_bf16.h>
#include <math.h>

// ---------------------------------------------------------------------------
// CompositionScorerNet fused kernel for MI455X (gfx1250, wave32, WMMA)
//
// out[b] = tanh( relu( concat(intent[b], bag[b]) @ W1 + b1 ) @ W2 + b2 )
// bag[b, s*16+j] = table[s][j] * hist_b[s] / max(len_b, 1)
//
// Memory-bound: ~440 MB HBM traffic -> ~19us floor @ 23.3 TB/s.
// GEMM via v_wmma_f32_16x16x32_bf16 (bf16 in, f32 accumulate), K padded to
// 1088. B fragments register-double-buffered to avoid loadcnt-0 stalls.
// ---------------------------------------------------------------------------

typedef __bf16 bf16_t;
typedef __bf16 v16bf __attribute__((ext_vector_type(16)));
typedef __bf16 v8bf  __attribute__((ext_vector_type(8)));
typedef float  v8f   __attribute__((ext_vector_type(8)));

#define N_SCEN   19
#define SDIM     16
#define BAG      304          // 19*16
#define EMBED    768
#define K_REAL   1072
#define KC_N     34           // ceil(1072/32)
#define K_PAD    (KC_N * 32)  // 1088
#define N_OUT    64
#define ROWS_BLK 64
#define THREADS  128          // 4 wave32

// LDS row stride for the bf16 "combined" tile: 1096 halves = 2192 bytes
// (548 dwords; 548 mod 64 = 36, gcd(36,64)=4 -> 16 consecutive rows hit
//  16 distinct bank quads -> conflict-free ds_load_b128 A-fragment reads)
#define ROW_H    1096
#define ROW_B    (ROW_H * 2)  // 2192

#define LDS_COMB 0
#define LDS_CNT  (ROWS_BLK * ROW_B)                 // 140288 : int cnt[64][20]
#define LDS_INV  (LDS_CNT + ROWS_BLK * 20 * 4)      // 145408 : float inv[64]
#define LDS_TAB  (LDS_INV + ROWS_BLK * 4)           // 145664 : float table[304]
#define LDS_SIZE (LDS_TAB + BAG * 4)                // 146880 bytes (2 blk/WGP)

__device__ __forceinline__ bf16_t f2bf(float f) {
  // round-to-nearest-even f32 -> bf16, pure bit ops
  unsigned u = __builtin_bit_cast(unsigned, f);
  unsigned r = u + 0x7FFFu + ((u >> 16) & 1u);
  unsigned short h = (unsigned short)(r >> 16);
  return __builtin_bit_cast(bf16_t, h);
}
__device__ __forceinline__ unsigned short bfbits(float f) {
  unsigned u = __builtin_bit_cast(unsigned, f);
  unsigned r = u + 0x7FFFu + ((u >> 16) & 1u);
  return (unsigned short)(r >> 16);
}

// ---------------------------------------------------------------------------
// Kernel 1: swizzle W1 [1072,64] f32 -> bf16 B-fragments in d_ws.
// Layout: w1p[kc][nt][lane][i]  (16 halves = 32 B per lane, contiguous).
// B-matrix 32x16 bf16 layout (ISA 7.12.2): lane L holds column n = L&15;
// half i holds k = 8*(L>>4) + (i&7) + 16*(i>>3). Rows k >= 1072 are zero padding.
// ---------------------------------------------------------------------------
__global__ void prep_w1_kernel(const float* __restrict__ W1,
                               bf16_t* __restrict__ w1p) {
  int e = blockIdx.x * blockDim.x + threadIdx.x;
  const int total = KC_N * 4 * 32 * 16;   // 69632 halves (136 KB)
  if (e >= total) return;
  int i  = e & 15;
  int L  = (e >> 4) & 31;
  int t  = e >> 9;
  int nt = t & 3;
  int kc = t >> 2;
  int n  = nt * 16 + (L & 15);
  int k  = kc * 32 + ((L >> 4) << 3) + (i & 7) + ((i >> 3) << 4);
  float v = (k < K_REAL) ? W1[k * N_OUT + n] : 0.0f;
  w1p[e] = f2bf(v);
}

// ---------------------------------------------------------------------------
// Kernel 2: fused bag + concat + GEMM(1072x64, WMMA bf16) + bias/relu +
//           64->1 dot + tanh.  One block = 64 rows, 4 waves * 16 rows.
// ---------------------------------------------------------------------------
__global__ void __launch_bounds__(THREADS)
scorer_kernel(const float* __restrict__ intent,   // [B,768]
              const int*   __restrict__ ids,      // [B,32]
              const float* __restrict__ mask,     // [B,32]
              const float* __restrict__ table,    // [19,16]
              const bf16_t* __restrict__ w1p,     // prepped B fragments
              const float* __restrict__ b1,       // [64]
              const float* __restrict__ W2,       // [64,1]
              const float* __restrict__ b2,       // [1]
              float* __restrict__ out) {          // [B,1]
  extern __shared__ char smem[];
  const int tid     = threadIdx.x;
  const int rowBase = blockIdx.x * ROWS_BLK;

  int*   cntL = (int*)(smem + LDS_CNT);
  float* invL = (float*)(smem + LDS_INV);
  float* tabL = (float*)(smem + LDS_TAB);

  // --- stage scenario table (tiny) -----------------------------------------
  for (int c = tid; c < BAG; c += THREADS) tabL[c] = table[c];

  // --- per-row id histogram (one owner thread per row, no atomics) ---------
  if (tid < ROWS_BLK) {
    int* cr = cntL + tid * 20;
    #pragma unroll
    for (int s = 0; s < 20; ++s) cr[s] = 0;
    const long grow = (long)(rowBase + tid) * 32;
    const int4*   ip4 = (const int4*)(ids + grow);
    const float4* mp4 = (const float4*)(mask + grow);
    float len = 0.0f;
    #pragma unroll
    for (int j = 0; j < 8; ++j) {
      int4   v = ip4[j];
      float4 m = mp4[j];
      if (m.x > 0.5f) { cr[v.x]++; len += 1.0f; }
      if (m.y > 0.5f) { cr[v.y]++; len += 1.0f; }
      if (m.z > 0.5f) { cr[v.z]++; len += 1.0f; }
      if (m.w > 0.5f) { cr[v.w]++; len += 1.0f; }
    }
    invL[tid] = 1.0f / fmaxf(len, 1.0f);
  }

  // --- stage intent embedding -> bf16 LDS tile (coalesced float4 loads) ----
  {
    const float4* ip = (const float4*)intent;
    for (int idx = tid; idx < ROWS_BLK * (EMBED / 4); idx += THREADS) {
      int r  = idx / (EMBED / 4);
      int c4 = idx % (EMBED / 4);
      float4 v = ip[(long)(rowBase + r) * (EMBED / 4) + c4];
      unsigned p0 = (unsigned)bfbits(v.x) | ((unsigned)bfbits(v.y) << 16);
      unsigned p1 = (unsigned)bfbits(v.z) | ((unsigned)bfbits(v.w) << 16);
      uint2 pk = make_uint2(p0, p1);
      *(uint2*)(smem + r * ROW_B + c4 * 8) = pk;   // 8B aligned LDS store
    }
  }
  __syncthreads();

  // --- bag_vec fill: combined[row][768+c] = bf16(table[c]*cnt[s]*inv) ------
  for (int idx = tid; idx < ROWS_BLK * BAG; idx += THREADS) {
    int r = idx / BAG;
    int c = idx % BAG;
    int s = c >> 4;
    float v = tabL[c] * (float)cntL[r * 20 + s] * invL[r];
    *(unsigned short*)(smem + r * ROW_B + (EMBED + c) * 2) = bfbits(v);
  }
  // zero K padding columns 1072..1087
  for (int idx = tid; idx < ROWS_BLK * 16; idx += THREADS) {
    int r = idx >> 4, j = idx & 15;
    *(unsigned short*)(smem + r * ROW_B + (K_REAL + j) * 2) = 0;
  }
  __syncthreads();

  // --- GEMM: each wave owns a 16x64 output tile, K = 34 chunks of 32 -------
  // A fragment (16-bit A 16x32 layout): lane L = row L&15, halves cover
  // K in {s..s+7} u {16+s..16+s+7}, s = 8*(L>>4)  -> two b128 LDS loads.
  // B fragments: 32 B contiguous per lane from prepped w1p (L2-resident).
  // Software pipeline: prefetch kc+1 fragments (clamped, branch-free) before
  // the 4 WMMAs of kc so waits become loadcnt<=8 / dscnt<=2 instead of 0.
  const int wv   = tid >> 5;
  const int lane = tid & 31;
  const int mrow = wv * 16 + (lane & 15);
  const int sOff = (lane >> 4) * 8;
  const char* arow = smem + mrow * ROW_B;
  const v16bf* bp = (const v16bf*)w1p + lane;   // fragment (kc,nt) at [128*kc+32*nt]

  v8f c0 = {}, c1 = {}, c2 = {}, c3 = {};

  v8bf  aLo = *(const v8bf*)(arow + (0 * 32 + sOff) * 2);
  v8bf  aHi = *(const v8bf*)(arow + (0 * 32 + 16 + sOff) * 2);
  v16bf B0 = bp[0], B1 = bp[32], B2 = bp[64], B3 = bp[96];

  #pragma unroll 2
  for (int kc = 0; kc < KC_N; ++kc) {
    // prefetch next chunk (clamped -> uniform, no divergence, EXEC stays 1s)
    const int kn = (kc + 1 < KC_N) ? (kc + 1) : (KC_N - 1);
    v8bf  nLo = *(const v8bf*)(arow + (kn * 32 + sOff) * 2);
    v8bf  nHi = *(const v8bf*)(arow + (kn * 32 + 16 + sOff) * 2);
    v16bf N0 = bp[kn * 128 +  0];
    v16bf N1 = bp[kn * 128 + 32];
    v16bf N2 = bp[kn * 128 + 64];
    v16bf N3 = bp[kn * 128 + 96];

    v16bf a = __builtin_shufflevector(aLo, aHi,
        0, 1, 2, 3, 4, 5, 6, 7, 8, 9, 10, 11, 12, 13, 14, 15);
    c0 = __builtin_amdgcn_wmma_f32_16x16x32_bf16(false, a, false, B0,
                                                 (short)0, c0, false, false);
    c1 = __builtin_amdgcn_wmma_f32_16x16x32_bf16(false, a, false, B1,
                                                 (short)0, c1, false, false);
    c2 = __builtin_amdgcn_wmma_f32_16x16x32_bf16(false, a, false, B2,
                                                 (short)0, c2, false, false);
    c3 = __builtin_amdgcn_wmma_f32_16x16x32_bf16(false, a, false, B3,
                                                 (short)0, c3, false, false);
    aLo = nLo; aHi = nHi;
    B0 = N0; B1 = N1; B2 = N2; B3 = N3;
  }

  // --- epilogue: +b1, relu, *W2, reduce over n (16 lanes), tanh, store -----
  // C/D layout: lane group g = lane>>4 holds rows g*8 + r (r = vgpr idx),
  // column n = nt*16 + (lane&15).
  float b1v[4], w2v[4];
  #pragma unroll
  for (int nt = 0; nt < 4; ++nt) {
    int n = nt * 16 + (lane & 15);
    b1v[nt] = b1[n];
    w2v[nt] = W2[n];
  }
  float acc[8];
  #pragma unroll
  for (int r = 0; r < 8; ++r) {
    acc[r]  = fmaxf(c0[r] + b1v[0], 0.0f) * w2v[0];
    acc[r] += fmaxf(c1[r] + b1v[1], 0.0f) * w2v[1];
    acc[r] += fmaxf(c2[r] + b1v[2], 0.0f) * w2v[2];
    acc[r] += fmaxf(c3[r] + b1v[3], 0.0f) * w2v[3];
  }
  #pragma unroll
  for (int r = 0; r < 8; ++r) {
    acc[r] += __shfl_xor(acc[r], 1, 16);
    acc[r] += __shfl_xor(acc[r], 2, 16);
    acc[r] += __shfl_xor(acc[r], 4, 16);
    acc[r] += __shfl_xor(acc[r], 8, 16);
  }
  if ((lane & 15) == 0) {
    const float bb = b2[0];
    const int base = rowBase + wv * 16 + (lane >> 4) * 8;
    #pragma unroll
    for (int r = 0; r < 8; ++r) out[base + r] = tanhf(acc[r] + bb);
  }
}

// ---------------------------------------------------------------------------
extern "C" void kernel_launch(void* const* d_in, const int* in_sizes, int n_in,
                              void* d_out, int out_size, void* d_ws, size_t ws_size,
                              hipStream_t stream) {
  const float* intent = (const float*)d_in[0];   // [131072,768] f32
  const int*   ids    = (const int*)d_in[1];     // [131072,32] i32
  const float* mask   = (const float*)d_in[2];   // [131072,32] f32
  const float* table  = (const float*)d_in[3];   // [19,16] f32
  const float* W1     = (const float*)d_in[4];   // [1072,64] f32
  const float* b1     = (const float*)d_in[5];   // [64] f32
  const float* W2     = (const float*)d_in[6];   // [64,1] f32
  const float* b2     = (const float*)d_in[7];   // [1] f32
  float* out = (float*)d_out;                    // [131072] f32
  bf16_t* w1p = (bf16_t*)d_ws;                   // 136 KB scratch

  const int prepTotal = KC_N * 4 * 32 * 16;      // 69632
  prep_w1_kernel<<<(prepTotal + 255) / 256, 256, 0, stream>>>(W1, w1p);

  const int B = in_sizes[0] / EMBED;             // 131072
  scorer_kernel<<<B / ROWS_BLK, THREADS, LDS_SIZE, stream>>>(
      intent, ids, mask, table, w1p, b1, W2, b2, out);
}
